// FpnMlpFsCls_2_13795434955105
// MI455X (gfx1250) — matrix operand order repaired and checked by reading, not compiled, not run on hardware
//
#include <hip/hip_runtime.h>

// Problem dims (fixed by the reference)
#define RR 32
#define BB 2048
#define FF 512
#define HH 1024
#define CC 100

typedef __attribute__((ext_vector_type(16))) __bf16        bf16x16;
typedef __attribute__((ext_vector_type(8)))  float         f32x8;
typedef __attribute__((ext_vector_type(4)))  unsigned int  u32x4;

__device__ __forceinline__ unsigned short f2bf(float x) {
  union { float f; unsigned int u; } c; c.f = x;
  unsigned int r = c.u + 0x7FFFu + ((c.u >> 16) & 1u);   // RNE
  return (unsigned short)(r >> 16);
}
__device__ __forceinline__ float bf2f(unsigned short s) {
  union { float f; unsigned int u; } c; c.u = ((unsigned int)s) << 16;
  return c.f;
}

// ---------------------------------------------------------------- cast f32->bf16
__global__ __launch_bounds__(256)
void cast_f32_bf16(const float* __restrict__ src, unsigned short* __restrict__ dst,
                   long long n) {
  long long i = (long long)blockIdx.x * blockDim.x + threadIdx.x;
  long long stride = (long long)gridDim.x * blockDim.x;
  for (; i < n; i += stride) dst[i] = f2bf(src[i]);
}

// ---------------------------------------------------------------- fuzzy membership
// fuzzy[r,b,f] = exp(-(data[b,f]-proto[r,f])^2 / (2*var[r,f]^2))  -> bf16 [R*B, F]
__global__ __launch_bounds__(256)
void fuzzy_kernel(const float* __restrict__ data, const float* __restrict__ proto,
                  const float* __restrict__ var, unsigned short* __restrict__ out) {
  long long n = (long long)RR * BB * FF;
  long long i = (long long)blockIdx.x * blockDim.x + threadIdx.x;
  long long stride = (long long)gridDim.x * blockDim.x;
  for (; i < n; i += stride) {
    int f = (int)(i % FF);
    long long t = i / FF;
    int b = (int)(t % BB);
    int r = (int)(t / BB);
    float d = data[(size_t)b * FF + f];
    float p = proto[(size_t)r * FF + f];
    float v = var[(size_t)r * FF + f];
    float x = d - p;
    out[i] = f2bf(__expf(-x * x / (2.0f * v * v)));
  }
}

// ---------------------------------------------------------------- WMMA GEMM
// out[bz] = act( A[bz] (MxK, bf16 rowmajor) * W[bz] (KxN, bf16 rowmajor) + bias[bz] )
// Block tile 128x128, K-tile 64. 8 waves (2x4), each wave 64x32 (4x2 frags).
// Requires K % 64 == 0, M % 128 == 0 (true for all call sites). N may be ragged.
// A tile is staged with GLOBAL_LOAD_ASYNC_TO_LDS_B128 (ASYNCcnt path);
// B tile goes through VGPRs because it is transposed into LDS ([n][k]) so that
// both WMMA fragment reads are contiguous ds_load_b128.
__global__ __launch_bounds__(256)
void gemm_bf16_wmma(const unsigned short* __restrict__ A, long long aStride,
                    const unsigned short* __restrict__ W, long long wStride,
                    const float* __restrict__ bias, long long biasStride,
                    void* __restrict__ out, long long oStride,
                    int M, int N, int K, int doRelu, int outF32) {
  __shared__ unsigned short lA[128 * 72];   // [row][k], k-stride 72 (pad)
  __shared__ unsigned short lB[128 * 72];   // transposed: [n][k], k-stride 72

  const int tid  = threadIdx.x;
  const int lane = tid & 31;
  const int wave = tid >> 5;
  const int wm   = (wave >> 2) * 64;   // wave M offset inside block tile
  const int wn   = (wave & 3) * 32;    // wave N offset inside block tile
  const int bz    = blockIdx.z;
  const int rBase = blockIdx.y * 128;
  const int nBase = blockIdx.x * 128;
  const unsigned short* Ab = A + (size_t)bz * aStride;
  const unsigned short* Wb = W + (size_t)bz * wStride;
  const int hlf  = lane >> 4;          // which 16-lane half
  const int lrow = lane & 15;

  union CF { f32x8 v; float f[8]; };
  CF acc[4][2];
#pragma unroll
  for (int i = 0; i < 4; i++)
#pragma unroll
    for (int j = 0; j < 2; j++)
#pragma unroll
      for (int e = 0; e < 8; e++) acc[i][j].f[e] = 0.0f;

  const bool vecW = ((N & 7) == 0);

  for (int kk = 0; kk < K; kk += 64) {
    // ---- stage A tile 128x64 (bf16): direct async global->LDS (no VGPR bounce).
    // M % 128 == 0 at all call sites, so no row guard is needed (EXEC all-ones).
#pragma unroll
    for (int i = 0; i < 4; i++) {
      int c    = tid + 256 * i;
      int row  = c >> 3;
      int kcol = (c & 7) * 8;
      int gr   = rBase + row;
      const unsigned short* gp = Ab + (size_t)gr * K + kk + kcol;
      unsigned int ldsoff = (unsigned int)(unsigned long long)(&lA[row * 72 + kcol]);
      asm volatile("global_load_async_to_lds_b128 %0, %1, off"
                   :: "v"(ldsoff), "v"((unsigned long long)gp)
                   : "memory");
    }
    // ---- stage B tile 64x128 transposed into lB[n][k]
#pragma unroll
    for (int i = 0; i < 4; i++) {
      int c    = tid + 256 * i;
      int krow = c >> 4;            // 0..63
      int ncol = (c & 15) * 8;
      int gk   = kk + krow;
      union { u32x4 q; unsigned short s[8]; } v;
      if (vecW && (nBase + ncol + 8) <= N) {
        v.q = *(const u32x4*)(Wb + (size_t)gk * N + nBase + ncol);
      } else {
#pragma unroll
        for (int j = 0; j < 8; j++) {
          int gc = nBase + ncol + j;
          v.s[j] = (gc < N) ? Wb[(size_t)gk * N + gc] : (unsigned short)0;
        }
      }
#pragma unroll
      for (int j = 0; j < 8; j++)
        lB[(ncol + j) * 72 + krow] = v.s[j];
    }
    // prefetch next W K-slab while we compute this one
    if (kk + 64 < K)
      __builtin_prefetch(Wb + (size_t)(kk + 64 + (tid >> 4)) * N + nBase + (tid & 15) * 8, 0, 1);
    // async A-tile copies must land in LDS before the barrier
    asm volatile("s_wait_asynccnt 0x0" ::: "memory");
    __syncthreads();

#pragma unroll
    for (int kf = 0; kf < 64; kf += 32) {
      union AFr { bf16x16 v; u32x4 q[2]; } afr[4], bfr[2];
      // A frag (16x32 bf16): lane half h, row lrow: k = kf + h*8 + {0..7, 16..23}
#pragma unroll
      for (int mi = 0; mi < 4; mi++) {
        const unsigned short* p = &lA[(wm + mi * 16 + lrow) * 72 + kf + hlf * 8];
        afr[mi].q[0] = *(const u32x4*)p;
        afr[mi].q[1] = *(const u32x4*)(p + 16);
      }
      // B frag (32x16 bf16): lane col lrow, k = kf + hlf*8 + {0..7, 16..23}
#pragma unroll
      for (int ni = 0; ni < 2; ni++) {
        const unsigned short* p = &lB[(wn + ni * 16 + lrow) * 72 + kf + hlf * 8];
        bfr[ni].q[0] = *(const u32x4*)p;
        bfr[ni].q[1] = *(const u32x4*)(p + 16);
      }
#pragma unroll
      for (int mi = 0; mi < 4; mi++)
#pragma unroll
        for (int ni = 0; ni < 2; ni++)
          acc[mi][ni].v = __builtin_amdgcn_wmma_f32_16x16x32_bf16(
              false, afr[mi].v, false, bfr[ni].v,
              (short)0, acc[mi][ni].v, false, false);
    }
    __syncthreads();
  }

  // ---- epilogue: C/D layout: col = nfrag + (lane&15); row = mfrag + e + 8*(lane>>4)
  const int mTop = hlf * 8;
#pragma unroll
  for (int mi = 0; mi < 4; mi++) {
#pragma unroll
    for (int ni = 0; ni < 2; ni++) {
      int col = nBase + wn + ni * 16 + lrow;
      float bv = 0.0f;
      if (bias && col < N) bv = bias[(size_t)bz * biasStride + col];
#pragma unroll
      for (int e = 0; e < 8; e++) {
        int row = rBase + wm + mi * 16 + mTop + e;
        if (row < M && col < N) {
          float v = acc[mi][ni].f[e] + bv;
          if (doRelu) v = v > 0.0f ? v : 0.0f;
          size_t idx = (size_t)bz * oStride + (size_t)row * N + col;
          if (outF32) ((float*)out)[idx] = v;
          else        ((unsigned short*)out)[idx] = f2bf(v);
        }
      }
    }
  }
}

// ---------------------------------------------------------------- fire-strength logit
// fs_logits[row] = dot(h2[row,:], fs_w3) + fs_b3, row in [0, R*B). One wave per row.
__global__ __launch_bounds__(256)
void fslogits_kernel(const unsigned short* __restrict__ h2,
                     const float* __restrict__ w3, const float* __restrict__ b3,
                     float* __restrict__ outRB) {
  int lane = threadIdx.x & 31;
  int row  = blockIdx.x * 8 + (threadIdx.x >> 5);
  const unsigned short* hp = h2 + (size_t)row * FF;
  float s = 0.0f;
  for (int f = lane; f < FF; f += 32) s += bf2f(hp[f]) * w3[f];
  for (int off = 16; off > 0; off >>= 1) s += __shfl_xor(s, off, 32);
  if (lane == 0) outRB[row] = s + b3[0];
}

// ---------------------------------------------------------------- softmax over rules
// fire[b,r] = softmax_r(fs_logits[r,b]).  Wave32: one lane per rule (R==32).
__global__ __launch_bounds__(256)
void fire_kernel(const float* __restrict__ fsl, float* __restrict__ fire) {
  int lane = threadIdx.x & 31;               // rule index
  int b = blockIdx.x * 8 + (threadIdx.x >> 5);
  float x = fsl[(size_t)lane * BB + b];
  float m = x;
  for (int off = 16; off > 0; off >>= 1) m = fmaxf(m, __shfl_xor(m, off, 32));
  float e = __expf(x - m);
  float s = e;
  for (int off = 16; off > 0; off >>= 1) s += __shfl_xor(s, off, 32);
  fire[(size_t)b * RR + lane] = e / s;
}

// ---------------------------------------------------------------- expert mix + class softmax
__global__ __launch_bounds__(128)
void final_kernel(const float* __restrict__ fire, const float* __restrict__ logits,
                  float* __restrict__ out) {
  __shared__ float red[128];
  int b = blockIdx.x;
  int c = threadIdx.x;
  float acc = 0.0f;
  if (c < CC) {
#pragma unroll 4
    for (int r = 0; r < RR; r++)
      acc += fire[(size_t)b * RR + r] * logits[((size_t)r * BB + b) * CC + c];
  }
  red[c] = (c < CC) ? acc : -3.4e38f;
  __syncthreads();
  for (int s = 64; s > 0; s >>= 1) { if (c < s) red[c] = fmaxf(red[c], red[c + s]); __syncthreads(); }
  float m = red[0];
  __syncthreads();
  float e = (c < CC) ? __expf(acc - m) : 0.0f;
  red[c] = e;
  __syncthreads();
  for (int s = 64; s > 0; s >>= 1) { if (c < s) red[c] += red[c + s]; __syncthreads(); }
  float ssum = red[0];
  if (c < CC) out[(size_t)b * CC + c] = e / ssum;
}

// ================================================================ launcher
extern "C" void kernel_launch(void* const* d_in, const int* in_sizes, int n_in,
                              void* d_out, int out_size, void* d_ws, size_t ws_size,
                              hipStream_t stream) {
  const float* data  = (const float*)d_in[0];
  const float* proto = (const float*)d_in[1];
  const float* var   = (const float*)d_in[2];
  const float* fs_w1 = (const float*)d_in[3];
  const float* fs_b1 = (const float*)d_in[4];
  const float* fs_w2 = (const float*)d_in[5];
  const float* fs_b2 = (const float*)d_in[6];
  const float* fs_w3 = (const float*)d_in[7];
  const float* fs_b3 = (const float*)d_in[8];
  const float* cq_w1 = (const float*)d_in[9];
  const float* cq_b1 = (const float*)d_in[10];
  const float* cq_w2 = (const float*)d_in[11];
  const float* cq_b2 = (const float*)d_in[12];
  const float* cq_w3 = (const float*)d_in[13];
  const float* cq_b3 = (const float*)d_in[14];
  float* out = (float*)d_out;   // [B, C] f32

  char* ws = (char*)d_ws;
  size_t off = 0;
  auto alloc = [&](size_t bytes) -> char* {
    char* p = ws + off;
    off += (bytes + 255) & ~(size_t)255;
    return p;
  };
  unsigned short* data_b  = (unsigned short*)alloc((size_t)BB * FF * 2);
  unsigned short* fs_w1b  = (unsigned short*)alloc((size_t)FF * HH * 2);
  unsigned short* fs_w2b  = (unsigned short*)alloc((size_t)HH * FF * 2);
  unsigned short* cq_w1b  = (unsigned short*)alloc((size_t)RR * FF * HH * 2);
  unsigned short* cq_w2b  = (unsigned short*)alloc((size_t)RR * HH * FF * 2);
  unsigned short* cq_w3b  = (unsigned short*)alloc((size_t)RR * FF * CC * 2);
  unsigned short* fuzzy   = (unsigned short*)alloc((size_t)RR * BB * FF * 2); // reused as g2
  unsigned short* h1      = (unsigned short*)alloc((size_t)RR * BB * HH * 2); // reused as g1
  unsigned short* h2      = (unsigned short*)alloc((size_t)RR * BB * FF * 2); // reused as logits(f32)
  float*          fsl     = (float*)alloc((size_t)RR * BB * 4);
  float*          fire    = (float*)alloc((size_t)BB * RR * 4);
  unsigned short* g1      = h1;
  unsigned short* g2      = fuzzy;
  float*          logits  = (float*)h2;      // 26.2 MB fits in the 64 MB h2 slab

  auto cg = [](long long n) {
    long long g = (n + 255) / 256;
    if (g > 16384) g = 16384;
    return (unsigned)g;
  };

  // 1) casts
  cast_f32_bf16<<<cg((long long)BB * FF), 256, 0, stream>>>(data, data_b, (long long)BB * FF);
  cast_f32_bf16<<<cg((long long)FF * HH), 256, 0, stream>>>(fs_w1, fs_w1b, (long long)FF * HH);
  cast_f32_bf16<<<cg((long long)HH * FF), 256, 0, stream>>>(fs_w2, fs_w2b, (long long)HH * FF);
  cast_f32_bf16<<<cg((long long)RR * FF * HH), 256, 0, stream>>>(cq_w1, cq_w1b, (long long)RR * FF * HH);
  cast_f32_bf16<<<cg((long long)RR * HH * FF), 256, 0, stream>>>(cq_w2, cq_w2b, (long long)RR * HH * FF);
  cast_f32_bf16<<<cg((long long)RR * FF * CC), 256, 0, stream>>>(cq_w3, cq_w3b, (long long)RR * FF * CC);

  // 2) fuzzy memberships  [R*B, F] bf16
  fuzzy_kernel<<<cg((long long)RR * BB * FF), 256, 0, stream>>>(data, proto, var, fuzzy);

  // 3) h1 = relu(fuzzy @ fs_w1 + fs_b1)   M=R*B, K=F, N=H
  gemm_bf16_wmma<<<dim3(HH / 128, (RR * BB) / 128, 1), 256, 0, stream>>>(
      fuzzy, 0, fs_w1b, 0, fs_b1, 0, h1, 0, RR * BB, HH, FF, 1, 0);

  // 4) h2 = relu(h1 @ fs_w2 + fs_b2)      M=R*B, K=H, N=F
  gemm_bf16_wmma<<<dim3(FF / 128, (RR * BB) / 128, 1), 256, 0, stream>>>(
      h1, 0, fs_w2b, 0, fs_b2, 0, h2, 0, RR * BB, FF, HH, 1, 0);

  // 5) fs_logits[r,b]
  fslogits_kernel<<<(RR * BB) / 8, 256, 0, stream>>>(h2, fs_w3, fs_b3, fsl);

  // 6) fire = softmax over rules
  fire_kernel<<<BB / 8, 256, 0, stream>>>(fsl, fire);

  // 7) g1 = relu(data @ cq_w1[r] + cq_b1[r])   grouped over r (A shared: stride 0)
  gemm_bf16_wmma<<<dim3(HH / 128, BB / 128, RR), 256, 0, stream>>>(
      data_b, 0, cq_w1b, (long long)FF * HH, cq_b1, HH,
      g1, (long long)BB * HH, BB, HH, FF, 1, 0);

  // 8) g2 = relu(g1 @ cq_w2[r] + cq_b2[r])
  gemm_bf16_wmma<<<dim3(FF / 128, BB / 128, RR), 256, 0, stream>>>(
      g1, (long long)BB * HH, cq_w2b, (long long)HH * FF, cq_b2, FF,
      g2, (long long)BB * FF, BB, FF, HH, 1, 0);

  // 9) logits = g2 @ cq_w3[r] + cq_b3[r]   (f32 out, no relu, ragged N=100)
  gemm_bf16_wmma<<<dim3((CC + 127) / 128, BB / 128, RR), 256, 0, stream>>>(
      g2, (long long)BB * FF, cq_w3b, (long long)FF * CC, cq_b3, CC,
      logits, (long long)BB * CC, BB, CC, FF, 0, 1);

  // 10) out = softmax_c( sum_r fire[b,r] * logits[r,b,c] )
  final_kernel<<<BB, 128, 0, stream>>>(fire, logits, out);
}
